// EfficientALU_L11_L12_7945689497952
// MI455X (gfx1250) — matrix operand order repaired and checked by reading, not compiled, not run on hardware
//
#include <hip/hip_runtime.h>

typedef __attribute__((ext_vector_type(16))) _Float16 v16h;
typedef __attribute__((ext_vector_type(8)))  _Float16 v8h;
typedef __attribute__((ext_vector_type(8)))  float    v8f;
typedef __attribute__((ext_vector_type(4)))  float    v4f;

#define GE        160
#define WSTRIDE   25600       // 160*160
#define TOK_PER_WAVE 32       // 4 strips of 16 rows (2 rows per token)
#define ROWS      64
#define NSTRIP    4
#define NT        10          // N tiles of 16 (160)
#define KC        5           // K chunks of 32 (160)
#define FRAG_HALVES 512       // 32 lanes * 16 halves per fragment

// ---------------------------------------------------------------------------
// Prep: convert dense layers 1..5 of W (f32, [7,160,160], h@W^T semantics) into
// f16 B-fragments in WMMA lane order so the main kernel does contiguous loads.
// B is KxN with B[k,n] = W[e = nt*16+n, d = kc*32+k].
// Lane layout (16-bit B, K=32): lanes 0-15 hold K {0..7,16..23}, lanes 16-31 K {8..15,24..31}.
// ---------------------------------------------------------------------------
__global__ void prep_weights_kernel(const float* __restrict__ W,
                                    _Float16* __restrict__ wsw) {
    int idx = blockIdx.x * blockDim.x + threadIdx.x;
    const int total = 5 * NT * KC * FRAG_HALVES;
    if (idx >= total) return;
    int i    = idx & 15;
    int lane = (idx >> 4) & 31;
    int kc   = (idx >> 9) % KC;
    int nt   = (idx / (FRAG_HALVES * KC)) % NT;
    int lp   = idx / (FRAG_HALVES * KC * NT);   // 0..4 -> dense layer lp+1
    int e = nt * 16 + (lane & 15);
    int d = kc * 32 + ((lane >> 4) << 3) + (i & 7) + ((i >> 3) << 4);
    wsw[idx] = (_Float16)W[(size_t)(lp + 1) * WSTRIDE + (size_t)e * GE + d];
}

// ---------------------------------------------------------------------------
// Main kernel: one wave = 32 tokens = 64 GE rows (only positions 0,1 matter).
// Each B fragment feeds 4 A-strips (4 WMMAs per weight load -> halves L2
// weight traffic). layer0 sparse (VALU), layers 1..5 dense f16 WMMA with LDS
// ping-pong for the D-layout -> A-layout rearrangement, layer6 reduced to a
// 160-dot (col RESULT=2), then fused non-temporal x_bd copy + masked scatter.
// ---------------------------------------------------------------------------
__global__ __launch_bounds__(32)
void alu_net_kernel(const float* __restrict__ x, const float* __restrict__ W,
                    const float* __restrict__ b, const _Float16* __restrict__ wsw,
                    float* __restrict__ out) {
    __shared__ __align__(16) _Float16 acts[2][ROWS * GE];   // 2 x 20480 B
    __shared__ float s_nibA[ROWS], s_nibB[ROWS], s_op[TOK_PER_WAVE];
    __shared__ int   s_mask[TOK_PER_WAVE];

    const int lane = threadIdx.x & 31;
    const int tok0 = blockIdx.x * TOK_PER_WAVE;

    // ---- decode: lane owns token t = lane (4 one-hot fields + gates/op) ----
    {
        int t = lane;
        size_t base = (size_t)(tok0 + t) * 512;
        float alo = 0.f, ahi = 0.f, blo = 0.f, bhi = 0.f;  // first k>0.5 else 0
        for (int k = 15; k >= 0; --k) {
            if (x[base +  64 + k] > 0.5f) alo = (float)k;  // ALU_LO
            if (x[base +  80 + k] > 0.5f) ahi = (float)k;  // ALU_HI
            if (x[base +  96 + k] > 0.5f) blo = (float)k;  // AX_CARRY_LO
            if (x[base + 112 + k] > 0.5f) bhi = (float)k;  // AX_CARRY_HI
        }
        s_nibA[t * 2 + 0] = alo;  s_nibA[t * 2 + 1] = ahi;
        s_nibB[t * 2 + 0] = blo;  s_nibB[t * 2 + 1] = bhi;
        s_op[t]   = x[base + 29];                              // OPCODE_BASE+27
        s_mask[t] = (x[base + 0] >= 0.5f && x[base + 1] >= 0.5f) ? 1 : 0;
    }
    __syncthreads();

    // ---- layer 0 (sparse input: cols {0,1,30}): lane owns rows lane, lane+32
    {
        int r0 = lane, r1 = lane + 32;
        float na0 = s_nibA[r0], nb0 = s_nibB[r0], op0 = s_op[r0 >> 1];
        float na1 = s_nibA[r1], nb1 = s_nibB[r1], op1 = s_op[r1 >> 1];
        _Float16* sm = &acts[0][0];
        for (int e = 0; e < GE; ++e) {
            const float* wr = W + (size_t)e * GE;   // layer 0 row e (uniform)
            float w0 = wr[0], w1 = wr[1], w30 = wr[30], be = b[e];
            float h0 = fmaxf(w0 * na0 + w1 * nb0 + w30 * op0 + be, 0.f);
            float h1 = fmaxf(w0 * na1 + w1 * nb1 + w30 * op1 + be, 0.f);
            sm[r0 * GE + e] = (_Float16)h0;
            sm[r1 * GE + e] = (_Float16)h1;
        }
    }
    __syncthreads();

    // ---- dense layers 1..5: WMMA f32 <- f16 x f16, 4 strips per B frag ----
    for (int l = 1; l <= 5; ++l) {
        const int ib = (l - 1) & 1, ob = l & 1;
        const _Float16* smi = &acts[ib][0];
        _Float16*       smo = &acts[ob][0];
        const _Float16* wl  = wsw + (size_t)(l - 1) * NT * KC * FRAG_HALVES;
        const int n  = lane & 15;
        const int hb = (lane >> 4) ? 8 : 0;     // A-layout K sub-offset
        for (int nt = 0; nt < NT; ++nt) {
            float bv = b[l * GE + nt * 16 + n];
            v8f acc[NSTRIP];
#pragma unroll
            for (int s = 0; s < NSTRIP; ++s)
                acc[s] = (v8f){bv, bv, bv, bv, bv, bv, bv, bv};
#pragma unroll
            for (int kc = 0; kc < KC; ++kc) {
                // B fragment: contiguous 32B per lane, L2/L0-resident
                v16h bf = *(const v16h*)(wl + ((size_t)(nt * KC + kc) * 32 + lane) * 16);
                int ab = kc * 32 + hb;
#pragma unroll
                for (int s = 0; s < NSTRIP; ++s) {
                    // A fragment (16-bit A layout): lanes 0-15 K {0..7,16..23},
                    // lanes 16-31 K {8..15,24..31}
                    const _Float16* p = smi + (s * 16 + n) * GE + ab;
                    v8h lo = *(const v8h*)p;
                    v8h hi = *(const v8h*)(p + 16);
                    v16h af = __builtin_shufflevector(lo, hi,
                        0,1,2,3,4,5,6,7,8,9,10,11,12,13,14,15);
                    acc[s] = __builtin_amdgcn_wmma_f32_16x16x32_f16(
                        false, af, false, bf, (short)0, acc[s], false, false);
                }
            }
            // ReLU + store in D layout: lane holds col n, rows M0..M0+7
            int M0 = (lane >> 4) * 8;
#pragma unroll
            for (int s = 0; s < NSTRIP; ++s)
#pragma unroll
                for (int v = 0; v < 8; ++v)
                    smo[(s * 16 + M0 + v) * GE + nt * 16 + n] =
                        (_Float16)fmaxf(acc[s][v], 0.f);
        }
        __syncthreads();
    }

    // ---- layer 6: only column RESULT=2 is used -> 160-dot per row ----
    int res0, res1;   // registers: lane L scatters rows L and L+32 itself
    {
        const _Float16* p0 = &acts[1][(size_t)lane * GE];        // l=5 wrote buf 1
        const _Float16* p1 = &acts[1][(size_t)(lane + 32) * GE];
        const float* w6 = W + (size_t)6 * WSTRIDE + 2 * GE;      // W[6][2][:]
        float a0 = b[6 * GE + 2], a1 = a0;
        for (int j = 0; j < GE; j += 8) {
            v8h h0 = *(const v8h*)(p0 + j);
            v8h h1 = *(const v8h*)(p1 + j);
#pragma unroll
            for (int q = 0; q < 8; ++q) {
                float wv = w6[j + q];        // uniform
                a0 += (float)h0[q] * wv;
                a1 += (float)h1[q] * wv;
            }
        }
        float r0 = fminf(fmaxf(rintf(a0), 0.f), 15.f);
        float r1 = fminf(fmaxf(rintf(a1), 0.f), 15.f);
        res0 = (int)r0;
        res1 = (int)r1;
    }

    // ---- fused output: non-temporal copy of 32 tokens + masked scatter ----
    {
        size_t base = (size_t)tok0 * 512;
        const v4f* src4 = (const v4f*)(x + base);
        v4f*       dst4 = (v4f*)(out + base);
        // 32 tokens * 512 f32 = 4096 float4; NT hints keep weights in L2
        for (int i = 0; i < 128; ++i) {
            v4f v = __builtin_nontemporal_load(&src4[i * 32 + lane]);
            __builtin_nontemporal_store(v, &dst4[i * 32 + lane]);
        }
        // row lane: token lane>>1, pos lane&1 ; row lane+32: token 16+(lane>>1)
        {
            int t = lane >> 1, pos = lane & 1;
            if (s_mask[t]) {
                size_t o = base + (size_t)t * 512 + (pos ? 144 : 128) + res0;
                out[o] = x[o] + 2.0f;          // overwrite after copy (in order)
            }
        }
        {
            int t = 16 + (lane >> 1), pos = lane & 1;
            if (s_mask[t]) {
                size_t o = base + (size_t)t * 512 + (pos ? 144 : 128) + res1;
                out[o] = x[o] + 2.0f;
            }
        }
    }
}

extern "C" void kernel_launch(void* const* d_in, const int* in_sizes, int n_in,
                              void* d_out, int out_size, void* d_ws, size_t ws_size,
                              hipStream_t stream) {
    const float* x = (const float*)d_in[0];   // [4,8192,512] f32
    const float* W = (const float*)d_in[1];   // [7,160,160]  f32
    const float* b = (const float*)d_in[2];   // [7,160]      f32
    float* out = (float*)d_out;
    _Float16* wsw = (_Float16*)d_ws;          // 256 KB fragment-ordered f16 weights

    const int total = 5 * NT * KC * FRAG_HALVES;
    prep_weights_kernel<<<(total + 255) / 256, 256, 0, stream>>>(W, wsw);

    // 32768 tokens / 32 per wave = 1024 waves; 1 wave per block
    alu_net_kernel<<<1024, 32, 0, stream>>>(x, W, b, wsw, out);
}
